// MultiHeadAttention_25477746000469
// MI455X (gfx1250) — matrix operand order repaired and checked by reading, not compiled, not run on hardware
//
#include <hip/hip_runtime.h>
#include <hip/hip_bf16.h>
#include <stdint.h>

#define BB  2
#define SS  2048
#define HH  1024
#define NHH 16
#define HDD 64
#define MT  (BB * SS)   // 4096 tokens

typedef __attribute__((ext_vector_type(16))) __bf16 bf16x16;
typedef __attribute__((ext_vector_type(8)))  __bf16 bf16x8;
typedef __attribute__((ext_vector_type(8)))  float  f32x8;

// ---------------------------------------------------------------------------
// Fragment loaders (layouts per CDNA5 ISA 7.12.2, wave32), vectorized b128.
// A: 16(M)x32(K) bf16. lane half -> Kbase 0/8; elems 0..7 -> K=kb+e,
//    elems 8..15 -> K=kb+16+e.
// ---------------------------------------------------------------------------
__device__ __forceinline__ bf16x16 loadA_f32(const float* __restrict__ src, int ld,
                                             int row0, int k0, int lane) {
  int half = lane >> 4;
  int r    = lane & 15;
  const float* p = src + (size_t)(row0 + r) * ld + k0 + half * 8;
  float4 x0 = *(const float4*)(p);
  float4 x1 = *(const float4*)(p + 4);
  float4 x2 = *(const float4*)(p + 16);
  float4 x3 = *(const float4*)(p + 20);
  bf16x16 a;
  a[0] = (__bf16)x0.x; a[1] = (__bf16)x0.y; a[2] = (__bf16)x0.z; a[3] = (__bf16)x0.w;
  a[4] = (__bf16)x1.x; a[5] = (__bf16)x1.y; a[6] = (__bf16)x1.z; a[7] = (__bf16)x1.w;
  a[8]  = (__bf16)x2.x; a[9]  = (__bf16)x2.y; a[10] = (__bf16)x2.z; a[11] = (__bf16)x2.w;
  a[12] = (__bf16)x3.x; a[13] = (__bf16)x3.y; a[14] = (__bf16)x3.z; a[15] = (__bf16)x3.w;
  return a;
}

__device__ __forceinline__ bf16x16 loadA_bf16(const __bf16* __restrict__ src, int ld,
                                              int row0, int k0, int lane) {
  int half = lane >> 4;
  int r    = lane & 15;
  const __bf16* p = src + (size_t)(row0 + r) * ld + k0 + half * 8;
  bf16x8 lo = *(const bf16x8*)(p);
  bf16x8 hi = *(const bf16x8*)(p + 16);
  return __builtin_shufflevector(lo, hi, 0, 1, 2, 3, 4, 5, 6, 7,
                                 8, 9, 10, 11, 12, 13, 14, 15);
}

// B: 32(K)x16(N). math B[k][n] = W[n][k] (W row-major, ld). lane -> column n,
// lane half -> Kbase 0/8; elems 0..7 -> K=kb+e, elems 8..15 -> K=kb+16+e.
__device__ __forceinline__ bf16x16 loadBT_f32(const float* __restrict__ w, int ld,
                                              int n0, int k0, int lane) {
  int n  = lane & 15;
  int kb = (lane >> 4) * 8;
  const float* p = w + (size_t)(n0 + n) * ld + k0 + kb;
  float4 x0 = *(const float4*)(p);
  float4 x1 = *(const float4*)(p + 4);
  float4 x2 = *(const float4*)(p + 16);
  float4 x3 = *(const float4*)(p + 20);
  bf16x16 b;
  b[0] = (__bf16)x0.x; b[1] = (__bf16)x0.y; b[2] = (__bf16)x0.z; b[3] = (__bf16)x0.w;
  b[4] = (__bf16)x1.x; b[5] = (__bf16)x1.y; b[6] = (__bf16)x1.z; b[7] = (__bf16)x1.w;
  b[8]  = (__bf16)x2.x; b[9]  = (__bf16)x2.y; b[10] = (__bf16)x2.z; b[11] = (__bf16)x2.w;
  b[12] = (__bf16)x3.x; b[13] = (__bf16)x3.y; b[14] = (__bf16)x3.z; b[15] = (__bf16)x3.w;
  return b;
}

__device__ __forceinline__ bf16x16 loadBT_bf16(const __bf16* __restrict__ w, int ld,
                                               int n0, int k0, int lane) {
  int n  = lane & 15;
  int kb = (lane >> 4) * 8;
  const __bf16* p = w + (size_t)(n0 + n) * ld + k0 + kb;
  bf16x8 lo = *(const bf16x8*)(p);
  bf16x8 hi = *(const bf16x8*)(p + 16);
  return __builtin_shufflevector(lo, hi, 0, 1, 2, 3, 4, 5, 6, 7,
                                 8, 9, 10, 11, 12, 13, 14, 15);
}

__device__ __forceinline__ f32x8 wmma_bf16(bf16x16 a, bf16x16 b, f32x8 c) {
  return __builtin_amdgcn_wmma_f32_16x16x32_bf16(false, a, false, b, (short)0, c,
                                                 false, false);
}

// ---------------------------------------------------------------------------
// Kernel 1: fused QKV projection.  X[4096,1024] @ W^T + b, outputs bf16:
//   Q -> [b,h,s,hd] scaled by 1/sqrt(HD);  K -> [b,h,s,hd];  V -> [b,h,hd,s]
// ---------------------------------------------------------------------------
__global__ void __launch_bounds__(256)
k_qkv(const float* __restrict__ q, const float* __restrict__ kk,
      const float* __restrict__ v,
      const float* __restrict__ Wq, const float* __restrict__ bq,
      const float* __restrict__ Wk, const float* __restrict__ bk,
      const float* __restrict__ Wv, const float* __restrict__ bv,
      __bf16* __restrict__ Qw, __bf16* __restrict__ Kw,
      __bf16* __restrict__ Vtw) {
  const int ntn = HH / 16;                       // 64 N tiles
  const int tiles_per_proj = (MT / 16) * ntn;    // 16384
  int wid = blockIdx.x * (blockDim.x >> 5) + (threadIdx.x >> 5);
  int p   = wid / tiles_per_proj;
  int t   = wid % tiles_per_proj;
  int m0  = (t / ntn) * 16;
  int n0  = (t % ntn) * 16;
  int lane = threadIdx.x & 31;

  const float* X    = (p == 0) ? q  : (p == 1) ? kk : v;
  const float* W    = (p == 0) ? Wq : (p == 1) ? Wk : Wv;
  const float* bias = (p == 0) ? bq : (p == 1) ? bk : bv;

  f32x8 acc = {};
  for (int k0 = 0; k0 < HH; k0 += 32) {
    __builtin_prefetch(X + (size_t)m0 * HH + k0 + 64, 0, 0);
    bf16x16 a = loadA_f32(X, HH, m0, k0, lane);
    bf16x16 b = loadBT_f32(W, HH, n0, k0, lane);
    acc = wmma_bf16(a, b, acc);
  }

  int n    = n0 + (lane & 15);
  int half = lane >> 4;
  float bsv   = bias[n];
  float scale = (p == 0) ? 0.125f : 1.0f;  // fold 1/sqrt(64) into Q
  int h = n >> 6, hd = n & 63;
#pragma unroll
  for (int r = 0; r < 8; ++r) {
    int m  = m0 + r + half * 8;
    int b_ = m >> 11;            // / 2048
    int s_ = m & (SS - 1);
    __bf16 val = (__bf16)((acc[r] + bsv) * scale);
    if (p == 0)      Qw[(((size_t)(b_ * NHH + h)) * SS + s_) * HDD + hd] = val;
    else if (p == 1) Kw[(((size_t)(b_ * NHH + h)) * SS + s_) * HDD + hd] = val;
    else             Vtw[(((size_t)(b_ * NHH + h)) * HDD + hd) * SS + s_] = val;
  }
}

// ---------------------------------------------------------------------------
// Kernel 2: fused scores + mask + softmax -> attn weights (fp32, written once).
// One block = one (b,h, 16-row strip). 8 waves x 256 columns = full S row.
// The whole K[b,h] panel (2048x64 bf16 = 256KB) is staged into LDS once per
// block with gfx1250 async loads, then consumed via ds_load B-fragments.
// ---------------------------------------------------------------------------
__global__ void __launch_bounds__(256)
k_attn(const __bf16* __restrict__ Qw, const __bf16* __restrict__ Kw,
       const int* __restrict__ mask, float* __restrict__ attn) {
  extern __shared__ char smem_raw[];
  __bf16* Klds = (__bf16*)smem_raw;      // [SS][HDD] = 256 KB

  int bh   = blockIdx.x / (SS / 16);
  int m0   = (blockIdx.x % (SS / 16)) * 16;
  int wave = threadIdx.x >> 5;
  int lane = threadIdx.x & 31;
  int half = lane >> 4;
  int nlane = lane & 15;
  int b_   = bh / NHH;
  int n0w  = wave * 256;

  const __bf16* Qb = Qw + (size_t)bh * SS * HDD;
  const __bf16* Kb = Kw + (size_t)bh * SS * HDD;
  const int* maskb = mask + (size_t)b_ * SS * SS;
  float* outb      = attn + (size_t)bh * SS * SS;

  // ---- async-stage K panel: 256KB / (256 thr * 16B) = 64 b128 ops/thread
  {
    uint32_t lbase = (uint32_t)(uintptr_t)Klds;
    const char* gK = (const char*)Kb;
#pragma unroll 4
    for (int i = 0; i < 64; ++i) {
      uint32_t off = (uint32_t)((i * 256 + threadIdx.x) * 16);
      uint32_t la  = lbase + off;
      uint64_t ga  = (uint64_t)(uintptr_t)(gK + off);
      asm volatile("global_load_async_to_lds_b128 %0, %1, off"
                   :: "v"(la), "v"(ga)
                   : "memory");
    }
    asm volatile("s_wait_asynccnt 0x0" ::: "memory");
    __syncthreads();
  }

  f32x8 acc[16];
#pragma unroll
  for (int t = 0; t < 16; ++t) acc[t] = (f32x8){};

  for (int k0 = 0; k0 < HDD; k0 += 32) {
    bf16x16 a = loadA_bf16(Qb, HDD, m0, k0, lane);
#pragma unroll
    for (int t = 0; t < 16; ++t) {
      bf16x16 b = loadBT_bf16(Klds, HDD, n0w + t * 16, k0, lane);
      acc[t] = wmma_bf16(a, b, acc[t]);
    }
  }

  __shared__ float wmax[8][16];
  __shared__ float wsum[8][16];
  __shared__ float rowred[16];
  __shared__ float sumred[16];

  // mask + per-lane row max
  float lmax[8];
#pragma unroll
  for (int r = 0; r < 8; ++r) lmax[r] = -3.0e38f;
#pragma unroll
  for (int t = 0; t < 16; ++t) {
    int j = n0w + t * 16 + nlane;
#pragma unroll
    for (int r = 0; r < 8; ++r) {
      int i = m0 + r + half * 8;
      float s = acc[t][r];
      if (maskb[(size_t)i * SS + j] == 0) s = -1.0e9f;
      acc[t][r] = s;
      lmax[r] = fmaxf(lmax[r], s);
    }
  }
  // reduce over the 16 lanes of each half-wave
#pragma unroll
  for (int r = 0; r < 8; ++r) {
    float m = lmax[r];
    for (int off = 1; off < 16; off <<= 1) m = fmaxf(m, __shfl_xor(m, off, 16));
    lmax[r] = m;
  }
  if (nlane == 0) {
#pragma unroll
    for (int r = 0; r < 8; ++r) wmax[wave][r + half * 8] = lmax[r];
  }
  __syncthreads();
  if (threadIdx.x < 16) {
    float m = wmax[0][threadIdx.x];
    for (int w = 1; w < 8; ++w) m = fmaxf(m, wmax[w][threadIdx.x]);
    rowred[threadIdx.x] = m;
  }
  __syncthreads();

  float rmax[8], lsum[8];
#pragma unroll
  for (int r = 0; r < 8; ++r) { rmax[r] = rowred[r + half * 8]; lsum[r] = 0.f; }
#pragma unroll
  for (int t = 0; t < 16; ++t) {
#pragma unroll
    for (int r = 0; r < 8; ++r) {
      float e = __expf(acc[t][r] - rmax[r]);
      acc[t][r] = e;
      lsum[r] += e;
    }
  }
#pragma unroll
  for (int r = 0; r < 8; ++r) {
    float s = lsum[r];
    for (int off = 1; off < 16; off <<= 1) s += __shfl_xor(s, off, 16);
    lsum[r] = s;
  }
  if (nlane == 0) {
#pragma unroll
    for (int r = 0; r < 8; ++r) wsum[wave][r + half * 8] = lsum[r];
  }
  __syncthreads();
  if (threadIdx.x < 16) {
    float s = 0.f;
    for (int w = 0; w < 8; ++w) s += wsum[w][threadIdx.x];
    sumred[threadIdx.x] = s;
  }
  __syncthreads();

  float rinv[8];
#pragma unroll
  for (int r = 0; r < 8; ++r) rinv[r] = 1.0f / sumred[r + half * 8];
#pragma unroll
  for (int t = 0; t < 16; ++t) {
    int j = n0w + t * 16 + nlane;
#pragma unroll
    for (int r = 0; r < 8; ++r) {
      int i = m0 + r + half * 8;
      outb[(size_t)i * SS + j] = acc[t][r] * rinv[r];
    }
  }
}

// ---------------------------------------------------------------------------
// Kernel 3: context = attn @ V.  attn fp32 [S,S] (from d_out), V^T bf16 [hd,s].
// Output ctx bf16 in token-major [b*s, H] layout for the final projection.
// ---------------------------------------------------------------------------
__global__ void __launch_bounds__(256)
k_ctx(const float* __restrict__ attn, const __bf16* __restrict__ Vtw,
      __bf16* __restrict__ ctx) {
  const int ntn = HDD / 16;  // 4
  int wid = blockIdx.x * (blockDim.x >> 5) + (threadIdx.x >> 5);
  int bh  = wid / ((SS / 16) * ntn);
  int t   = wid % ((SS / 16) * ntn);
  int m0  = (t / ntn) * 16;
  int n0  = (t % ntn) * 16;
  int lane = threadIdx.x & 31;

  const float* Ab  = attn + (size_t)bh * SS * SS;
  const __bf16* Vb = Vtw + (size_t)bh * HDD * SS;

  f32x8 acc = {};
  for (int k0 = 0; k0 < SS; k0 += 32) {
    bf16x16 a = loadA_f32(Ab, SS, m0, k0, lane);
    bf16x16 b = loadBT_bf16(Vb, SS, n0, k0, lane);
    acc = wmma_bf16(a, b, acc);
  }

  int b_ = bh / NHH, h = bh % NHH;
  int half = lane >> 4;
  int hd = n0 + (lane & 15);
#pragma unroll
  for (int r = 0; r < 8; ++r) {
    int s_ = m0 + r + half * 8;
    ctx[((size_t)(b_ * SS + s_)) * HH + h * HDD + hd] = (__bf16)acc[r];
  }
}

// ---------------------------------------------------------------------------
// Kernel 4: output projection ctx[4096,1024](bf16) @ Wo^T + bo -> fp32 out
// ---------------------------------------------------------------------------
__global__ void __launch_bounds__(256)
k_oproj(const __bf16* __restrict__ ctx, const float* __restrict__ Wo,
        const float* __restrict__ bo, float* __restrict__ out) {
  const int ntn = HH / 16;  // 64
  int wid = blockIdx.x * (blockDim.x >> 5) + (threadIdx.x >> 5);
  int m0  = (wid / ntn) * 16;
  int n0  = (wid % ntn) * 16;
  int lane = threadIdx.x & 31;

  f32x8 acc = {};
  for (int k0 = 0; k0 < HH; k0 += 32) {
    bf16x16 a = loadA_bf16(ctx, HH, m0, k0, lane);
    bf16x16 b = loadBT_f32(Wo, HH, n0, k0, lane);
    acc = wmma_bf16(a, b, acc);
  }
  int n    = n0 + (lane & 15);
  int half = lane >> 4;
  float bsv = bo[n];
#pragma unroll
  for (int r = 0; r < 8; ++r) {
    int m = m0 + r + half * 8;
    out[(size_t)m * HH + n] = acc[r] + bsv;
  }
}

// ---------------------------------------------------------------------------
extern "C" void kernel_launch(void* const* d_in, const int* in_sizes, int n_in,
                              void* d_out, int out_size, void* d_ws, size_t ws_size,
                              hipStream_t stream) {
  const float* q    = (const float*)d_in[0];
  const float* k    = (const float*)d_in[1];
  const float* v    = (const float*)d_in[2];
  const int*   mask = (const int*)d_in[3];
  const float* Wq   = (const float*)d_in[4];
  const float* bq   = (const float*)d_in[5];
  const float* Wk   = (const float*)d_in[6];
  const float* bk   = (const float*)d_in[7];
  const float* Wv   = (const float*)d_in[8];
  const float* bv   = (const float*)d_in[9];
  const float* Wo   = (const float*)d_in[10];
  const float* bo   = (const float*)d_in[11];

  float* out  = (float*)d_out;                   // [B,S,H]
  float* attn = out + (size_t)BB * SS * HH;      // [B,NH,S,S]

  const size_t nQ = (size_t)BB * NHH * SS * HDD; // 4.19M elems
  __bf16* Qw  = (__bf16*)d_ws;
  __bf16* Kw  = Qw + nQ;
  __bf16* Vtw = Kw + nQ;
  __bf16* ctx = Vtw + nQ;                        // [B*S, H] bf16

  {
    int waves = 3 * (MT / 16) * (HH / 16);       // 49152
    k_qkv<<<dim3(waves / 8), dim3(256), 0, stream>>>(q, k, v, Wq, bq, Wk, bk, Wv,
                                                     bv, Qw, Kw, Vtw);
  }
  // 256KB dynamic LDS: full K head panel per block
  k_attn<<<dim3(BB * NHH * (SS / 16)), dim3(256), SS * HDD * sizeof(__hip_bfloat16),
           stream>>>(Qw, Kw, mask, attn);
  {
    int waves = BB * NHH * (SS / 16) * (HDD / 16);  // 16384
    k_ctx<<<dim3(waves / 8), dim3(256), 0, stream>>>(attn, Vtw, ctx);
  }
  {
    int waves = (MT / 16) * (HH / 16);              // 16384
    k_oproj<<<dim3(waves / 8), dim3(256), 0, stream>>>(ctx, Wo, bo, out);
  }
}
